// deformConv_50723563765973
// MI455X (gfx1250) — matrix operand order repaired and checked by reading, not compiled, not run on hardware
//
#include <hip/hip_runtime.h>

#define HH   256
#define WWID 256
#define HWs  (HH * WWID)
#define CINC 64
#define COUTC 64

typedef __attribute__((ext_vector_type(16))) __bf16 v16bf;
typedef __attribute__((ext_vector_type(8)))  float  v8f;

// round-to-nearest-even fp32 -> bf16 bit pattern
static __device__ __forceinline__ unsigned short f2bf_bits(float f) {
    unsigned u = __float_as_uint(f);
    unsigned r = u + 0x7FFFu + ((u >> 16) & 1u);
    return (unsigned short)(r >> 16);
}
static __device__ __forceinline__ float bf_bits2f(unsigned short s) {
    return __uint_as_float(((unsigned)s) << 16);
}

union FragU { uint4 q[2]; v16bf v; };

static __device__ __forceinline__ v16bf load_frag16(const unsigned short* p) {
    FragU f;
    f.q[0] = *(const uint4*)(p);
    f.q[1] = *(const uint4*)(p + 8);
    return f.v;
}

// Map weight element (m=cout row 0..63, k=cin 0..63) to this kernel's contiguous-K 16-bit
// A-matrix fragment layout (16x32 per fragment, 2 K-steps, 4 M-tiles):
//   lanes 0-15 : M=m%16, elements 0..15 hold K=0..15
//   lanes 16-31: same M, elements 0..15 hold K=16..31
static __device__ __forceinline__ int frag_off(int m, int k) {
    int mt = m >> 4, ml = m & 15;
    int ks = k >> 5, kl = k & 31;
    int hi = kl >> 4;              // K >= 16 within the 32-chunk
    int kk = kl & 15;
    int lane = ml + (hi << 4);
    int e = kk;                                    // element 0..15
    return ((mt * 2 + ks) * 32 + lane) * 16 + e;
}

// S tile: [256 px][72 shorts] (64 cin + pad to 144B row stride: 36-bank stride,
// 16 consecutive pixels land in disjoint bank quads -> conflict-free b128 loads)
#define SROW 72

__global__ __launch_bounds__(256)
void deform_conv_wmma(const float* __restrict__ x,
                      const float* __restrict__ offset,
                      const float* __restrict__ weight,
                      const float* __restrict__ bias,
                      float* __restrict__ out) {
    __shared__ __align__(16) unsigned short Shi[256 * SROW];
    __shared__ __align__(16) unsigned short Slo[256 * SROW];
    __shared__ __align__(32) unsigned short Whi[4096];   // current tap, frag layout
    __shared__ __align__(32) unsigned short Wlo[4096];

    const int tid  = threadIdx.x;
    const int lane = tid & 31;
    const int wv   = tid >> 5;                 // wave 0..7
    const int tx   = blockIdx.x, ty = blockIdx.y;

    // sampling: this thread's output pixel
    const int p  = tid;
    const int oy = ty * 16 + (p >> 4);
    const int ox = tx * 16 + (p & 15);

    v8f acc[4][2];
#pragma unroll
    for (int mt = 0; mt < 4; ++mt) { acc[mt][0] = {}; acc[mt][1] = {}; }

    for (int tap = 0; tap < 9; ++tap) {
        // ---- phase 1a: swizzle this tap's 64x64 weights into A-frag layout (split bf16)
#pragma unroll 4
        for (int it = 0; it < 16; ++it) {
            int idx = tid + (it << 8);             // 0..4095
            int m = idx >> 6, k = idx & 63;
            float wvv = weight[(m * 64 + k) * 9 + tap];
            unsigned short hb = f2bf_bits(wvv);
            unsigned short lb = f2bf_bits(wvv - bf_bits2f(hb));
            int fo = frag_off(m, k);
            Whi[fo] = hb;
            Wlo[fo] = lb;
        }

        // ---- phase 1b: bilinear gather -> split-bf16 im2col tile in LDS
        {
            const int ky = tap / 3, kx = tap % 3;
            float offy = offset[(tap * 2 + 0) * HWs + oy * WWID + ox];
            float offx = offset[(tap * 2 + 1) * HWs + oy * WWID + ox];
            if (tap < 8) {   // pull next tap's offset rows toward the caches
                __builtin_prefetch(offset + (tap * 2 + 2) * HWs + oy * WWID + ox, 0, 3);
                __builtin_prefetch(offset + (tap * 2 + 3) * HWs + oy * WWID + ox, 0, 3);
            }
            float fy = (float)(oy - 1 + ky) + offy;
            float fx = (float)(ox - 1 + kx) + offx;
            float y0f = floorf(fy), x0f = floorf(fx);
            float wy1 = fy - y0f, wx1 = fx - x0f;
            float wy0 = 1.0f - wy1, wx0 = 1.0f - wx1;
            int y0 = (int)y0f, x0 = (int)x0f;
            int y1 = y0 + 1, x1 = x0 + 1;
            bool vy0 = (unsigned)y0 < 256u, vy1 = (unsigned)y1 < 256u;
            bool vx0 = (unsigned)x0 < 256u, vx1 = (unsigned)x1 < 256u;
            float m00 = (vy0 && vx0) ? wy0 * wx0 : 0.0f;
            float m01 = (vy0 && vx1) ? wy0 * wx1 : 0.0f;
            float m10 = (vy1 && vx0) ? wy1 * wx0 : 0.0f;
            float m11 = (vy1 && vx1) ? wy1 * wx1 : 0.0f;
            int y0c = min(max(y0, 0), 255), y1c = min(max(y1, 0), 255);
            int x0c = min(max(x0, 0), 255), x1c = min(max(x1, 0), 255);
            int i00 = y0c * WWID + x0c, i01 = y0c * WWID + x1c;
            int i10 = y1c * WWID + x0c, i11 = y1c * WWID + x1c;
            unsigned short* sh = &Shi[p * SROW];
            unsigned short* sl = &Slo[p * SROW];
            const float* xp = x;
#pragma unroll 4
            for (int c = 0; c < CINC; ++c) {
                float s = xp[i00] * m00 + xp[i01] * m01 + xp[i10] * m10 + xp[i11] * m11;
                unsigned short hb = f2bf_bits(s);
                sh[c] = hb;
                sl[c] = f2bf_bits(s - bf_bits2f(hb));
                xp += HWs;
            }
        }
        __syncthreads();

        // ---- phase 2: WMMA — C(64 cout x 256 px) += W(64x64) * S(64x256), split-bf16
        {
            const int bl   = lane & 15;
            const int koff = (lane & 16);          // 0 or 16: B K-range per lane half
            const int col0 = wv * 32 + bl;         // wave owns pixels [wv*32, wv*32+32)
#pragma unroll
            for (int ks = 0; ks < 2; ++ks) {
                const int kb = ks * 32 + koff;
                v16bf b0h = load_frag16(&Shi[col0 * SROW + kb]);
                v16bf b0l = load_frag16(&Slo[col0 * SROW + kb]);
                v16bf b1h = load_frag16(&Shi[(col0 + 16) * SROW + kb]);
                v16bf b1l = load_frag16(&Slo[(col0 + 16) * SROW + kb]);
#pragma unroll
                for (int mt = 0; mt < 4; ++mt) {
                    v16bf ah = load_frag16(&Whi[((mt * 2 + ks) * 32 + lane) * 16]);
                    v16bf al = load_frag16(&Wlo[((mt * 2 + ks) * 32 + lane) * 16]);
                    acc[mt][0] = __builtin_amdgcn_wmma_f32_16x16x32_bf16(false, ah, false, b0h, (short)0, acc[mt][0], false, false);
                    acc[mt][0] = __builtin_amdgcn_wmma_f32_16x16x32_bf16(false, ah, false, b0l, (short)0, acc[mt][0], false, false);
                    acc[mt][0] = __builtin_amdgcn_wmma_f32_16x16x32_bf16(false, al, false, b0h, (short)0, acc[mt][0], false, false);
                    acc[mt][1] = __builtin_amdgcn_wmma_f32_16x16x32_bf16(false, ah, false, b1h, (short)0, acc[mt][1], false, false);
                    acc[mt][1] = __builtin_amdgcn_wmma_f32_16x16x32_bf16(false, ah, false, b1l, (short)0, acc[mt][1], false, false);
                    acc[mt][1] = __builtin_amdgcn_wmma_f32_16x16x32_bf16(false, al, false, b1h, (short)0, acc[mt][1], false, false);
                }
            }
        }
        __syncthreads();   // S/W reused next tap
    }

    // ---- epilogue: bias + coalesced store (C layout: lane=col, VGPR r=row, +8 for lanes>=16)
    {
        const int bl    = lane & 15;
        const int mhalf = (lane >> 4) << 3;     // 0 or 8
#pragma unroll
        for (int mt = 0; mt < 4; ++mt) {
#pragma unroll
            for (int r = 0; r < 8; ++r) {
                int cout = mt * 16 + r + mhalf;
                float bv = bias[cout];
#pragma unroll
                for (int nt = 0; nt < 2; ++nt) {
                    int pp  = (wv * 2 + nt) * 16 + bl;
                    int oyo = ty * 16 + (pp >> 4);
                    int oxo = tx * 16 + (pp & 15);
                    out[cout * HWs + oyo * WWID + oxo] = acc[mt][nt][r] + bv;
                }
            }
        }
    }
}

extern "C" void kernel_launch(void* const* d_in, const int* in_sizes, int n_in,
                              void* d_out, int out_size, void* d_ws, size_t ws_size,
                              hipStream_t stream) {
    (void)in_sizes; (void)n_in; (void)out_size; (void)d_ws; (void)ws_size;
    const float* x      = (const float*)d_in[0];
    const float* offset = (const float*)d_in[1];
    const float* weight = (const float*)d_in[2];
    const float* bias   = (const float*)d_in[3];
    float* out = (float*)d_out;
    dim3 grid(16, 16, 1);    // 256 tiles of 16x16 output pixels
    dim3 block(256, 1, 1);   // 8 wave32
    hipLaunchKernelGGL(deform_conv_wmma, grid, block, 0, stream,
                       x, offset, weight, bias, out);
}
